// Attention_90915867722430
// MI455X (gfx1250) — matrix operand order repaired and checked by reading, not compiled, not run on hardware
//
#include <hip/hip_runtime.h>
#include <hip/hip_bf16.h>

typedef __attribute__((ext_vector_type(16))) _Float16 v16h;
typedef __attribute__((ext_vector_type(8)))  _Float16 v8h;
typedef __attribute__((ext_vector_type(8)))  float    v8f;

#define HEADS    12
#define HEAD_DIM 64
#define EMBED    768
#define SEQ      2048
#define BATCH    2
#define QKV_OUT  2304   // 3*EMBED
#define MTOT     4096   // BATCH*SEQ
#define NKB      (SEQ / 32)

// ---------------------------------------------------------------------------
// D = A x B + C via v_wmma_f32_16x16x32_f16 (wave32)
// ---------------------------------------------------------------------------
__device__ __forceinline__ v8f wmma_f16(v16h a, v16h b, v8f c) {
  return __builtin_amdgcn_wmma_f32_16x16x32_f16(
      /*neg_a=*/false, a, /*neg_b=*/false, b,
      /*c_mod=*/(short)0, c, /*reuse_a=*/false, /*reuse_b=*/false);
}

// A-fragment 16x32 f16: lane l -> row (l&15); khalf=(l>>4):
//   elements 0..7  = K (khalf*16 .. khalf*16+7)
//   elements 8..15 = K (khalf*16+8 .. khalf*16+15)
__device__ __forceinline__ v16h load_a_frag(const _Float16* rowptr, int khalf) {
  v8h lo = *(const v8h*)(rowptr + khalf * 16);
  v8h hi = *(const v8h*)(rowptr + khalf * 16 + 8);
  v16h r;
#pragma unroll
  for (int i = 0; i < 8; ++i) { r[i] = lo[i]; r[i + 8] = hi[i]; }
  return r;
}

// gfx1250 async copy global -> LDS (16B per lane), tracked by ASYNCcnt.
// lds_off = AS3 byte offset (low 32 bits of a generic pointer to __shared__).
__device__ __forceinline__ void async_load_b128(unsigned lds_off,
                                                const _Float16* gptr) {
  unsigned long long ga = (unsigned long long)gptr;
  asm volatile("global_load_async_to_lds_b128 %0, %1, off"
               :
               : "v"(lds_off), "v"(ga)
               : "memory");
}

__device__ __forceinline__ unsigned lds_off_of(const void* p) {
  return (unsigned)(unsigned long long)p;  // low 32 bits = AS3 offset
}

// ---------------------------------------------------------------------------
// fp32 -> fp16 conversion (grid-stride)
// ---------------------------------------------------------------------------
__global__ __launch_bounds__(256) void f32_to_f16_kernel(
    const float* __restrict__ in, _Float16* __restrict__ out, int n) {
  int i = blockIdx.x * blockDim.x + threadIdx.x;
  int stride = gridDim.x * blockDim.x;
  for (; i < n; i += stride) out[i] = (_Float16)in[i];
}

// ---------------------------------------------------------------------------
// QKV GEMM: qkv[m,o] = sum_k x16[m,k] * Wqkv16[o,k]
// M=4096, K=768, N=2304. Wave computes a 16x64 tile (4 accumulators).
// Block = 256 thr = 8 waves -> 32M x 256N block tile.
// Output scattered to q16 (B,H,N,Dh), k16 (B,H,N,Dh), vT16 (B,H,Dh,N), f16.
// ---------------------------------------------------------------------------
__global__ __launch_bounds__(256) void qkv_gemm_kernel(
    const _Float16* __restrict__ X, const _Float16* __restrict__ W,
    _Float16* __restrict__ q16, _Float16* __restrict__ k16,
    _Float16* __restrict__ vT16) {
  const int lane  = threadIdx.x & 31;
  const int wave  = threadIdx.x >> 5;
  const int mlo   = lane & 15;
  const int khalf = lane >> 4;
  const int mbase = blockIdx.y * 32 + (wave & 1) * 16;
  const int nbase = blockIdx.x * 256 + (wave >> 1) * 64;

  v8f acc[4] = {};
  for (int k = 0; k < EMBED; k += 32) {
    const _Float16* ap = X + (size_t)(mbase + mlo) * EMBED + k;
    __builtin_prefetch(ap + 64, 0, 1);          // global_prefetch_b8
    v16h a = load_a_frag(ap, khalf);
#pragma unroll
    for (int t = 0; t < 4; ++t) {
      // B-frag: lane holds 16 contiguous K for column (nbase+t*16+mlo)
      const _Float16* bp = W + (size_t)(nbase + t * 16 + mlo) * EMBED + k + khalf * 16;
      __builtin_prefetch(bp + 64, 0, 1);
      v16h b = *(const v16h*)bp;
      acc[t] = wmma_f16(a, b, acc[t]);
    }
  }

#pragma unroll
  for (int t = 0; t < 4; ++t) {
    int o     = nbase + t * 16 + mlo;
    int which = o / EMBED;          // 0=q, 1=k, 2=v
    int rem   = o - which * EMBED;
    int h     = rem >> 6;
    int d     = rem & 63;
#pragma unroll
    for (int v = 0; v < 8; ++v) {
      int m  = mbase + v + 8 * khalf;
      int b  = m >> 11;             // /SEQ
      int n  = m & (SEQ - 1);
      int bh = b * HEADS + h;
      _Float16 val = (_Float16)acc[t][v];
      if (which == 0)      q16[((size_t)bh * SEQ + n) * HEAD_DIM + d] = val;
      else if (which == 1) k16[((size_t)bh * SEQ + n) * HEAD_DIM + d] = val;
      else                 vT16[((size_t)bh * HEAD_DIM + d) * SEQ + n] = val;
    }
  }
}

// ---------------------------------------------------------------------------
// Flash attention with double-buffered async K/V staging in LDS.
// One wave owns 16 query rows of one (b,h); the 4 waves of a block share the
// same (b,h) and stream the same 32-key blocks, so K (32x64 f16) and V
// (64x32 f16, transposed) tiles are fetched ONCE per block per iteration via
// global_load_async_to_lds_b128 (ASYNCcnt) and consumed via ds_load_b128.
// ---------------------------------------------------------------------------
__global__ __launch_bounds__(128) void attn_kernel(
    const _Float16* __restrict__ q16, const _Float16* __restrict__ k16,
    const _Float16* __restrict__ vT16, _Float16* __restrict__ O16) {
  __shared__ _Float16 Kbuf[2][32][64];   // [buf][key][d]      8 KB
  __shared__ _Float16 Vbuf[2][64][32];   // [buf][d][key]      8 KB
  __shared__ _Float16 Plds[4][16][32];   // per-wave P tile    4 KB

  const int tid   = threadIdx.x;
  const int lane  = tid & 31;
  const int wave  = tid >> 5;
  const int mlo   = lane & 15;
  const int khalf = lane >> 4;
  const int bh    = blockIdx.y;
  const int b     = bh / HEADS;
  const int h     = bh - b * HEADS;
  const int q0    = blockIdx.x * 64 + wave * 16;

  const _Float16* qp = q16 + (size_t)bh * SEQ * HEAD_DIM;
  const _Float16* kp = k16 + (size_t)bh * SEQ * HEAD_DIM;
  const _Float16* vp = vT16 + (size_t)bh * HEAD_DIM * SEQ;

  // Issue one K/V tile fetch: 512 x 16B chunks over 128 threads = 4 per thread
  auto issue_tile = [&](int kb, int buf) {
    const _Float16* gk = kp + (size_t)kb * 32 * HEAD_DIM;  // contiguous 4KB
#pragma unroll
    for (int j = 0; j < 2; ++j) {
      int c = tid + 128 * j;                               // 0..255
      async_load_b128(lds_off_of(&Kbuf[buf][0][0] + c * 8), gk + c * 8);
    }
#pragma unroll
    for (int j = 0; j < 2; ++j) {
      int cv = tid + 128 * j;                              // 0..255
      int d = cv >> 2, part = cv & 3;
      async_load_b128(lds_off_of(&Vbuf[buf][d][part * 8]),
                      vp + (size_t)d * SEQ + kb * 32 + part * 8);
    }
  };

  // Q fragments: two K=32 chunks covering Dh=64, resident for whole loop
  v16h aQ[2];
#pragma unroll
  for (int c = 0; c < 2; ++c)
    aQ[c] = load_a_frag(qp + (size_t)(q0 + mlo) * HEAD_DIM + c * 32, khalf);

  v8f acc[4] = {};
  float rowmax[8], rowsum[8];
#pragma unroll
  for (int v = 0; v < 8; ++v) { rowmax[v] = -1e30f; rowsum[v] = 0.0f; }

  const float scale = 0.125f;  // HEAD_DIM^-0.5

  issue_tile(0, 0);

  for (int kb = 0; kb < NKB; ++kb) {
    const int cur = kb & 1;
    if (kb + 1 < NKB) {
      issue_tile(kb + 1, cur ^ 1);
      // 4 newly issued per wave remain; the 4 for tile `cur` retire in order
      asm volatile("s_wait_asynccnt 0x4" ::: "memory");
    } else {
      asm volatile("s_wait_asynccnt 0x0" ::: "memory");
    }
    __syncthreads();   // K/V tile `cur` visible to all waves

    // ---- scores: S(16q x 32keys) as two 16x16 C fragments, K=Dh=64 ----
    v8f s[2] = {};
#pragma unroll
    for (int t = 0; t < 2; ++t) {
      const _Float16* krow = &Kbuf[cur][t * 16 + mlo][0];
      v16h b0 = *(const v16h*)(krow + khalf * 16);        // d = khalf*16..+15
      v16h b1 = *(const v16h*)(krow + 32 + khalf * 16);   // d = 32+khalf*16..
      s[t] = wmma_f16(aQ[0], b0, s[t]);
      s[t] = wmma_f16(aQ[1], b1, s[t]);
    }

    // ---- online softmax (row stats live per-vgpr, broadcast in 16-lane group)
#pragma unroll
    for (int v = 0; v < 8; ++v) {
      float s0 = s[0][v] * scale;
      float s1 = s[1][v] * scale;
      float m  = fmaxf(s0, s1);
#pragma unroll
      for (int msk = 1; msk < 16; msk <<= 1)
        m = fmaxf(m, __shfl_xor(m, msk, 32));
      float mnew = fmaxf(rowmax[v], m);
      float fac  = __expf(rowmax[v] - mnew);
      rowmax[v]  = mnew;
      float p0 = __expf(s0 - mnew);
      float p1 = __expf(s1 - mnew);
      float rs = p0 + p1;
#pragma unroll
      for (int msk = 1; msk < 16; msk <<= 1)
        rs += __shfl_xor(rs, msk, 32);
      rowsum[v] = rowsum[v] * fac + rs;
#pragma unroll
      for (int t = 0; t < 4; ++t) acc[t][v] *= fac;
      // C-layout -> LDS (row-major 16x32) for A-frag reload
      Plds[wave][v + 8 * khalf][mlo]      = (_Float16)p0;
      Plds[wave][v + 8 * khalf][mlo + 16] = (_Float16)p1;
    }
    __syncthreads();

    // ---- reload P as A-fragment (16x32) ----
    v16h aP;
    {
      v8h lo = *(const v8h*)(&Plds[wave][mlo][khalf * 16]);
      v8h hi = *(const v8h*)(&Plds[wave][mlo][khalf * 16 + 8]);
#pragma unroll
      for (int i = 0; i < 8; ++i) { aP[i] = lo[i]; aP[i + 8] = hi[i]; }
    }

    // ---- O += P @ V : transposed V tile gives contiguous B-frag ds loads ----
#pragma unroll
    for (int t = 0; t < 4; ++t) {
      v16h bV = *(const v16h*)(&Vbuf[cur][t * 16 + mlo][khalf * 16]);
      acc[t] = wmma_f16(aP, bV, acc[t]);
    }
    __syncthreads();   // all waves done with tile `cur` before it is re-filled
  }

  // ---- normalize + store O (B,N,EMBED) f16 ----
#pragma unroll
  for (int t = 0; t < 4; ++t) {
    int c = h * HEAD_DIM + t * 16 + mlo;
#pragma unroll
    for (int v = 0; v < 8; ++v) {
      int n = q0 + v + 8 * khalf;
      float o = acc[t][v] / rowsum[v];
      O16[((size_t)(b * SEQ + n)) * EMBED + c] = (_Float16)o;
    }
  }
}

// ---------------------------------------------------------------------------
// Output projection: out[m,o] = sum_k O16[m,k] * Wproj16[o,k], fp32 result.
// M=4096, K=768, N=768.
// ---------------------------------------------------------------------------
__global__ __launch_bounds__(256) void proj_gemm_kernel(
    const _Float16* __restrict__ A, const _Float16* __restrict__ W,
    float* __restrict__ out) {
  const int lane  = threadIdx.x & 31;
  const int wave  = threadIdx.x >> 5;
  const int mlo   = lane & 15;
  const int khalf = lane >> 4;
  const int mbase = blockIdx.y * 32 + (wave & 1) * 16;
  const int nbase = blockIdx.x * 256 + (wave >> 1) * 64;

  v8f acc[4] = {};
  for (int k = 0; k < EMBED; k += 32) {
    const _Float16* ap = A + (size_t)(mbase + mlo) * EMBED + k;
    __builtin_prefetch(ap + 64, 0, 1);
    v16h a = load_a_frag(ap, khalf);
#pragma unroll
    for (int t = 0; t < 4; ++t) {
      const _Float16* bp = W + (size_t)(nbase + t * 16 + mlo) * EMBED + k + khalf * 16;
      __builtin_prefetch(bp + 64, 0, 1);
      v16h b = *(const v16h*)bp;
      acc[t] = wmma_f16(a, b, acc[t]);
    }
  }
#pragma unroll
  for (int t = 0; t < 4; ++t) {
    int o = nbase + t * 16 + mlo;
#pragma unroll
    for (int v = 0; v < 8; ++v) {
      int m = mbase + v + 8 * khalf;
      out[(size_t)m * EMBED + o] = acc[t][v];
    }
  }
}

// ---------------------------------------------------------------------------
extern "C" void kernel_launch(void* const* d_in, const int* in_sizes, int n_in,
                              void* d_out, int out_size, void* d_ws, size_t ws_size,
                              hipStream_t stream) {
  const float* x     = (const float*)d_in[0];   // (2,2048,768)
  const float* Wqkv  = (const float*)d_in[1];   // (2304,768)
  const float* Wproj = (const float*)d_in[2];   // (768,768)
  float* out = (float*)d_out;                   // (2,2048,768)

  char* ws = (char*)d_ws;
  size_t off = 0;
  auto alloc = [&](size_t bytes) -> void* {
    void* p = ws + off;
    off = (off + bytes + 255) & ~(size_t)255;
    return p;
  };
  _Float16* x16  = (_Float16*)alloc((size_t)MTOT * EMBED * 2);
  _Float16* wq16 = (_Float16*)alloc((size_t)QKV_OUT * EMBED * 2);
  _Float16* wp16 = (_Float16*)alloc((size_t)EMBED * EMBED * 2);
  _Float16* q16  = (_Float16*)alloc((size_t)BATCH * HEADS * SEQ * HEAD_DIM * 2);
  _Float16* k16  = (_Float16*)alloc((size_t)BATCH * HEADS * SEQ * HEAD_DIM * 2);
  _Float16* vT16 = (_Float16*)alloc((size_t)BATCH * HEADS * HEAD_DIM * SEQ * 2);
  _Float16* O16  = (_Float16*)alloc((size_t)MTOT * EMBED * 2);

  f32_to_f16_kernel<<<2048, 256, 0, stream>>>(x, x16, MTOT * EMBED);
  f32_to_f16_kernel<<<2048, 256, 0, stream>>>(Wqkv, wq16, QKV_OUT * EMBED);
  f32_to_f16_kernel<<<1024, 256, 0, stream>>>(Wproj, wp16, EMBED * EMBED);

  qkv_gemm_kernel<<<dim3(QKV_OUT / 256, MTOT / 32), 256, 0, stream>>>(
      x16, wq16, q16, k16, vT16);

  attn_kernel<<<dim3(SEQ / 64, BATCH * HEADS), 128, 0, stream>>>(
      q16, k16, vT16, O16);

  proj_gemm_kernel<<<dim3(EMBED / 256, MTOT / 32), 256, 0, stream>>>(
      O16, wp16, out);
}
